// Transform_83167746720135
// MI455X (gfx1250) — matrix operand order repaired and checked by reading, not compiled, not run on hardware
//
#include <hip/hip_runtime.h>

// ---------------------------------------------------------------------------
// PointNet-ish forward for MI455X (gfx1250, wave32, WMMA).
// Heavy channel-convs (O x C x B*N GEMMs) run on v_wmma_f32_16x16x32_f16.
// ---------------------------------------------------------------------------

typedef __attribute__((ext_vector_type(16))) _Float16 v16h;
typedef __attribute__((ext_vector_type(8)))  float    v8f;
typedef __attribute__((ext_vector_type(2)))  __fp16   v2fp;

union Frag16 {
    v16h     v;
    _Float16 h[16];
    uint4    q[2];
};

union PackH2 {
    v2fp         v;
    unsigned int u;
};

static inline int cdiv(int a, int b) { return (a + b - 1) / b; }

// ---------------------------------------------------------------------------
// WMMA GEMM:  Y[b][o][n] = sum_c W[o][c] * X[b][c][n] + bias[o]
// Block: 128 threads (4 waves). Block tile: 64 (O) x 64 (N). K step: 32.
// Requires O % 64 == 0, C % 32 == 0, N % 64 == 0.
// Staging packs two f32 -> one b32 LDS store via v_cvt_pk_rtz_f16_f32.
// ---------------------------------------------------------------------------
__global__ __launch_bounds__(128)
void wmma_gemm_kernel(const float* __restrict__ W, const float* __restrict__ X,
                      const float* __restrict__ bias, float* __restrict__ Y,
                      int O, int C, int Nn)
{
    const int b    = blockIdx.z;
    const int o0   = blockIdx.y * 64;
    const int n0   = blockIdx.x * 64;
    const int tid  = threadIdx.x;
    const int lane = tid & 31;
    const int wv   = tid >> 5;

    const float* Xb = X + (size_t)b * C * Nn;
    float*       Yb = Y + (size_t)b * O * Nn;

    __shared__ __align__(16) _Float16 sA[64 * 32];  // [o_row][k]
    __shared__ __align__(16) _Float16 sB[64 * 32];  // transposed: [n_col][k]
    unsigned int* sA32 = (unsigned int*)sA;         // 16 dwords per row
    unsigned int* sB32 = (unsigned int*)sB;

    v8f acc[4] = {};

    for (int c0 = 0; c0 < C; c0 += 32) {
        if (c0 + 32 < C) {  // CDNA5 global_prefetch of next K chunk
            __builtin_prefetch(Xb + (size_t)(c0 + 32) * Nn + n0 + lane, 0, 1);
            __builtin_prefetch(W + (size_t)(o0 + (tid >> 1)) * C + c0 + 32, 0, 1);
        }
        // Stage A tile 64x32 as 1024 half-pairs; lane-major for coalescing.
        // pair t: row r = t>>4, k-pair kp = t&15 -> k = 2*kp (contiguous in W row).
        for (int i = 0; i < 8; ++i) {
            int t  = i * 128 + tid;
            int r  = t >> 4, kp = t & 15;
            const float* wp = &W[(size_t)(o0 + r) * C + c0 + 2 * kp];
            PackH2 p; p.v = __builtin_amdgcn_cvt_pkrtz(wp[0], wp[1]);
            sA32[r * 16 + kp] = p.u;
        }
        // Stage B tile transposed (sB[j][k]) as 1024 half-pairs.
        // pair t: kp = t>>6 (k = 2*kp), j = t&63; consecutive lanes -> consecutive j.
        for (int i = 0; i < 8; ++i) {
            int t  = i * 128 + tid;
            int kp = t >> 6, j = t & 63;
            const float* xp = &Xb[(size_t)(c0 + 2 * kp) * Nn + n0 + j];
            PackH2 p; p.v = __builtin_amdgcn_cvt_pkrtz(xp[0], xp[Nn]);
            sB32[j * 16 + kp] = p.u;
        }
        __syncthreads();

        // A fragment (16x32 f16). ISA layout: lanes 0-15 row M=lane hold
        // K {0..7,16..23}; lanes 16-31 row M=lane-16 hold K {8..15,24..31}.
        Frag16 af;
        {
            int row = wv * 16 + (lane & 15);
            int lo  = (lane < 16) ? 0 : 8;
            af.q[0] = *(const uint4*)&sA[row * 32 + lo];
            af.q[1] = *(const uint4*)&sA[row * 32 + lo + 16];
        }
        // Four 16x16 N sub-tiles.
        for (int nt = 0; nt < 4; ++nt) {
            Frag16 bf;
            int col = nt * 16 + (lane & 15);
            int kb  = (lane < 16) ? 0 : 8;    // B: same K split as A
            bf.q[0] = *(const uint4*)&sB[col * 32 + kb];
            bf.q[1] = *(const uint4*)&sB[col * 32 + kb + 16];
            acc[nt] = __builtin_amdgcn_wmma_f32_16x16x32_f16(
                false, af.v, false, bf.v, (short)0, acc[nt], false, false);
        }
        __syncthreads();
    }

    // Epilogue: 16x16 f32 C/D layout: lanes 0-15 N=lane, rows M=vgpr;
    // lanes 16-31 N=lane-16, rows M=vgpr+8. Fused bias add.
    for (int nt = 0; nt < 4; ++nt) {
        int col   = n0 + nt * 16 + (lane & 15);
        int rbase = o0 + wv * 16 + ((lane < 16) ? 0 : 8);
        for (int r = 0; r < 8; ++r) {
            int row = rbase + r;
            Yb[(size_t)row * Nn + col] = acc[nt][r] + bias[row];
        }
    }
}

// ---------------------------------------------------------------------------
// Small-C conv (C = 3 or 6): one thread per (b,o,n).
// ---------------------------------------------------------------------------
__global__ void conv_small_kernel(const float* __restrict__ W, const float* __restrict__ X,
                                  const float* __restrict__ bias, float* __restrict__ Y,
                                  int O, int C, int Nn, int Bn)
{
    int idx = blockIdx.x * blockDim.x + threadIdx.x;
    int total = Bn * O * Nn;
    if (idx >= total) return;
    int n = idx % Nn;
    int t = idx / Nn;
    int o = t % O;
    int b = t / O;
    const float* Xb = X + (size_t)b * C * Nn + n;
    float acc = bias[o];
    for (int c = 0; c < C; ++c) acc += W[o * C + c] * Xb[(size_t)c * Nn];
    Y[idx] = acc;
}

// ---------------------------------------------------------------------------
// BatchNorm over (B, N) per channel — stats pass (one block per channel).
// ---------------------------------------------------------------------------
__global__ __launch_bounds__(256)
void bn3d_stats_kernel(const float* __restrict__ X, float* __restrict__ mean,
                       float* __restrict__ inv, int Cc, int Nn, int Bn, float eps)
{
    int c = blockIdx.x;
    __shared__ float ss[256], sq[256];
    float s = 0.f, q = 0.f;
    for (int b = 0; b < Bn; ++b) {
        const float* p = X + ((size_t)b * Cc + c) * Nn;
        for (int n = threadIdx.x; n < Nn; n += 256) {
            float v = p[n];
            s += v; q += v * v;
        }
    }
    ss[threadIdx.x] = s; sq[threadIdx.x] = q;
    __syncthreads();
    for (int st = 128; st > 0; st >>= 1) {
        if (threadIdx.x < st) {
            ss[threadIdx.x] += ss[threadIdx.x + st];
            sq[threadIdx.x] += sq[threadIdx.x + st];
        }
        __syncthreads();
    }
    if (threadIdx.x == 0) {
        float cnt = (float)(Bn * Nn);
        float m   = ss[0] / cnt;
        float var = sq[0] / cnt - m * m;
        mean[c] = m;
        inv[c]  = rsqrtf(var + eps);
    }
}

__global__ void bn3d_apply_kernel(float* __restrict__ X, const float* __restrict__ mean,
                                  const float* __restrict__ inv, const float* __restrict__ g,
                                  const float* __restrict__ bb, int Cc, int Nn, int total, int relu)
{
    int idx = blockIdx.x * blockDim.x + threadIdx.x;
    if (idx >= total) return;
    int c = (idx / Nn) % Cc;
    float v = (X[idx] - mean[c]) * inv[c] * g[c] + bb[c];
    if (relu) v = fmaxf(v, 0.f);
    X[idx] = v;
}

// ---------------------------------------------------------------------------
// Max over N per (b,c). grid = B*C blocks.
// ---------------------------------------------------------------------------
__global__ __launch_bounds__(256)
void maxn_kernel(const float* __restrict__ X, float* __restrict__ out, int Nn)
{
    int bc = blockIdx.x;
    const float* p = X + (size_t)bc * Nn;
    __shared__ float sm[256];
    float m = -3.4e38f;
    for (int n = threadIdx.x; n < Nn; n += 256) m = fmaxf(m, p[n]);
    sm[threadIdx.x] = m;
    __syncthreads();
    for (int st = 128; st > 0; st >>= 1) {
        if (threadIdx.x < st) sm[threadIdx.x] = fmaxf(sm[threadIdx.x], sm[threadIdx.x + st]);
        __syncthreads();
    }
    if (threadIdx.x == 0) out[bc] = sm[0];
}

// ---------------------------------------------------------------------------
// FC: Y[b][o] = sum_k W[o][k] X[b][k] + bias[o]. One thread per (b,o).
// ---------------------------------------------------------------------------
__global__ void fc_kernel(const float* __restrict__ W, const float* __restrict__ X,
                          const float* __restrict__ bias, float* __restrict__ Y,
                          int O, int Kk, int Bn)
{
    int idx = blockIdx.x * blockDim.x + threadIdx.x;
    if (idx >= Bn * O) return;
    int o = idx % O, b = idx / O;
    const float* xr = X + (size_t)b * Kk;
    const float* wr = W + (size_t)o * Kk;
    float a = bias[o];
    for (int k = 0; k < Kk; ++k) a += wr[k] * xr[k];
    Y[idx] = a;
}

// BatchNorm over batch dim only (B=8). One thread per feature, in place.
__global__ void bn1d_kernel(float* __restrict__ X, const float* __restrict__ g,
                            const float* __restrict__ bb, int F, int Bn, float eps, int relu)
{
    int f = blockIdx.x * blockDim.x + threadIdx.x;
    if (f >= F) return;
    float s = 0.f, q = 0.f;
    for (int b = 0; b < Bn; ++b) {
        float v = X[(size_t)b * F + f];
        s += v; q += v * v;
    }
    float m   = s / Bn;
    float var = q / Bn - m * m;
    float inv = rsqrtf(var + eps);
    for (int b = 0; b < Bn; ++b) {
        float v = (X[(size_t)b * F + f] - m) * inv * g[f] + bb[f];
        if (relu) v = fmaxf(v, 0.f);
        X[(size_t)b * F + f] = v;
    }
}

__global__ void addeye_kernel(const float* __restrict__ X, float* __restrict__ Y, int k, int Bn)
{
    int idx = blockIdx.x * blockDim.x + threadIdx.x;
    if (idx >= Bn * k * k) return;
    int e = idx % (k * k);
    int r = e / k, c = e % k;
    Y[idx] = X[idx] + ((r == c) ? 1.f : 0.f);
}

// out[b,d,n] = sum_c X[b,c,n] * M[b,c,d]
__global__ void transform_kernel(const float* __restrict__ X, const float* __restrict__ M,
                                 float* __restrict__ Y, int Kk, int Nn)
{
    int n = blockIdx.x * blockDim.x + threadIdx.x;
    int d = blockIdx.y, b = blockIdx.z;
    if (n >= Nn) return;
    const float* Xb = X + (size_t)b * Kk * Nn;
    const float* Mb = M + (size_t)b * Kk * Kk;
    float a = 0.f;
    for (int c = 0; c < Kk; ++c) a += Xb[(size_t)c * Nn + n] * Mb[c * Kk + d];
    Y[((size_t)b * Kk + d) * Nn + n] = a;
}

// concat: ch 0..2 from xb3 (b,3,n); ch 3..5 from normals (b,n,3) transposed.
__global__ void concat6_kernel(const float* __restrict__ xb3, const float* __restrict__ nrm,
                               float* __restrict__ out, int Nn, int Bn)
{
    int idx = blockIdx.x * blockDim.x + threadIdx.x;
    int total = Bn * 6 * Nn;
    if (idx >= total) return;
    int n  = idx % Nn;
    int t  = idx / Nn;
    int ch = t % 6;
    int b  = t / 6;
    float v = (ch < 3) ? xb3[((size_t)b * 3 + ch) * Nn + n]
                       : nrm[((size_t)b * Nn + n) * 3 + (ch - 3)];
    out[idx] = v;
}

// ---------------------------------------------------------------------------
// Normal estimation: one thread per point. Top-20 NN (excl. self), centered
// covariance of diffs, smallest eigenvector of symmetric 3x3 (analytic).
// ---------------------------------------------------------------------------
__global__ __launch_bounds__(128)
void normals_kernel(const float* __restrict__ x /* (B,3,N) */,
                    float* __restrict__ nrm /* (B,N,3) */, int Nn, int Bn)
{
    int idx = blockIdx.x * blockDim.x + threadIdx.x;
    if (idx >= Bn * Nn) return;
    int i = idx % Nn, b = idx / Nn;
    const float* px = x + (size_t)b * 3 * Nn;
    float xi = px[i], yi = px[Nn + i], zi = px[2 * Nn + i];

    const int KN = 20;
    float bd[KN]; int bj[KN];
    for (int t = 0; t < KN; ++t) { bd[t] = 3.4e38f; bj[t] = 0; }
    for (int j = 0; j < Nn; ++j) {
        if (j == i) continue;
        float dx = xi - px[j], dy = yi - px[Nn + j], dz = zi - px[2 * Nn + j];
        float d2 = dx * dx + dy * dy + dz * dz;
        if (d2 < bd[KN - 1]) {
            int t = KN - 1;
            while (t > 0 && bd[t - 1] > d2) { bd[t] = bd[t - 1]; bj[t] = bj[t - 1]; --t; }
            bd[t] = d2; bj[t] = j;
        }
    }
    // Covariance of (p_i - p_j) around its mean: S = sum d d^T - KN * m m^T.
    float sx = 0, sy = 0, sz = 0;
    float sxx = 0, sxy = 0, sxz = 0, syy = 0, syz = 0, szz = 0;
    for (int t = 0; t < KN; ++t) {
        int j = bj[t];
        float dx = xi - px[j], dy = yi - px[Nn + j], dz = zi - px[2 * Nn + j];
        sx += dx; sy += dy; sz += dz;
        sxx += dx * dx; sxy += dx * dy; sxz += dx * dz;
        syy += dy * dy; syz += dy * dz; szz += dz * dz;
    }
    float kinv = 1.f / KN;
    float mx = sx * kinv, my = sy * kinv, mz = sz * kinv;
    float a00 = sxx - KN * mx * mx, a01 = sxy - KN * mx * my, a02 = sxz - KN * mx * mz;
    float a11 = syy - KN * my * my, a12 = syz - KN * my * mz, a22 = szz - KN * mz * mz;

    // Smallest eigenvalue of symmetric 3x3 (trigonometric formula).
    float q  = (a00 + a11 + a22) * (1.f / 3.f);
    float p1 = a01 * a01 + a02 * a02 + a12 * a12;
    float b00 = a00 - q, b11 = a11 - q, b22 = a22 - q;
    float p2 = b00 * b00 + b11 * b11 + b22 * b22 + 2.f * p1;
    float p  = sqrtf(fmaxf(p2 * (1.f / 6.f), 1e-30f));
    float ip = 1.f / p;
    float c00 = b00 * ip, c01 = a01 * ip, c02 = a02 * ip;
    float c11 = b11 * ip, c12 = a12 * ip, c22 = b22 * ip;
    float detB = c00 * (c11 * c22 - c12 * c12) - c01 * (c01 * c22 - c12 * c02)
               + c02 * (c01 * c12 - c11 * c02);
    float r   = fminf(fmaxf(detB * 0.5f, -1.f), 1.f);
    float phi = acosf(r) * (1.f / 3.f);
    float lam = q + 2.f * p * cosf(phi + 2.0943951023931953f);  // smallest

    // Eigenvector: largest cross product of rows of (S - lam I).
    float r0x = a00 - lam, r0y = a01,       r0z = a02;
    float r1x = a01,       r1y = a11 - lam, r1z = a12;
    float r2x = a02,       r2y = a12,       r2z = a22 - lam;
    float cx0 = r0y * r1z - r0z * r1y, cy0 = r0z * r1x - r0x * r1z, cz0 = r0x * r1y - r0y * r1x;
    float cx1 = r0y * r2z - r0z * r2y, cy1 = r0z * r2x - r0x * r2z, cz1 = r0x * r2y - r0y * r2x;
    float cx2 = r1y * r2z - r1z * r2y, cy2 = r1z * r2x - r1x * r2z, cz2 = r1x * r2y - r1y * r2x;
    float n0 = cx0 * cx0 + cy0 * cy0 + cz0 * cz0;
    float n1 = cx1 * cx1 + cy1 * cy1 + cz1 * cz1;
    float n2 = cx2 * cx2 + cy2 * cy2 + cz2 * cz2;
    float vx = cx0, vy = cy0, vz = cz0, nb = n0;
    if (n1 > nb) { vx = cx1; vy = cy1; vz = cz1; nb = n1; }
    if (n2 > nb) { vx = cx2; vy = cy2; vz = cz2; nb = n2; }
    float innv = rsqrtf(fmaxf(nb, 1e-30f));
    nrm[((size_t)b * Nn + i) * 3 + 0] = vx * innv;
    nrm[((size_t)b * Nn + i) * 3 + 1] = vy * innv;
    nrm[((size_t)b * Nn + i) * 3 + 2] = vz * innv;
}

// ---------------------------------------------------------------------------
// Host orchestration.
// Input order = jax pytree flatten (sorted dict keys):
//  0:x, 1:bn1.b 2:bn1.g 3:bn2.b 4:bn2.g 5:bn3.b 6:bn3.g
//  7:c1b 8:c1w 9:c2b 10:c2w 11:c3b 12:c3w
//  t3 @13, t64 @35, each ordered:
//   +0 b1 +1 b2 +2 b3 +3 bn1.b +4 bn1.g +5 bn2.b +6 bn2.g +7 bn3.b +8 bn3.g
//   +9 bn4.b +10 bn4.g +11 bn5.b +12 bn5.g +13 fb1 +14 fb2 +15 fb3
//   +16 fw1 +17 fw2 +18 fw3 +19 w1 +20 w2 +21 w3
// Output: out (8*1024) | m3 (8*9) | m64 (8*64*64)
// ---------------------------------------------------------------------------
extern "C" void kernel_launch(void* const* d_in, const int* in_sizes, int n_in,
                              void* d_out, int out_size, void* d_ws, size_t ws_size,
                              hipStream_t stream)
{
    (void)in_sizes; (void)n_in; (void)out_size; (void)ws_size;
    const int Bn = 8, Nn = 4096;
    const float* x = (const float*)d_in[0];
    auto P = [&](int i) { return (const float*)d_in[i]; };

    float* ws = (float*)d_ws;
    size_t off = 0;
    auto alloc = [&](size_t n) { float* p = ws + off; off += n; return p; };
    float* bufA   = alloc((size_t)Bn * 64 * Nn);
    float* bufB   = alloc((size_t)Bn * 128 * Nn);
    float* bufC   = alloc((size_t)Bn * 1024 * Nn);
    float* bufD   = alloc((size_t)Bn * 64 * Nn);   // xb (post-c1)
    float* bufE   = alloc((size_t)Bn * 64 * Nn);   // xb @ m64
    float* xb3    = alloc((size_t)Bn * 3 * Nn);
    float* nrm    = alloc((size_t)Bn * Nn * 3);
    float* xb6    = alloc((size_t)Bn * 6 * Nn);
    float* pooled = alloc((size_t)Bn * 1024);
    float* f1     = alloc((size_t)Bn * 512);
    float* f2     = alloc((size_t)Bn * 256);
    float* f3     = alloc((size_t)Bn * 4096);
    float* meanb  = alloc(1024);
    float* invb   = alloc(1024);

    float* out_max = (float*)d_out;
    float* m3      = out_max + Bn * 1024;
    float* m64     = m3 + Bn * 9;

    auto bn3d = [&](float* X, int Cc, const float* g, const float* bb, int relu) {
        bn3d_stats_kernel<<<Cc, 256, 0, stream>>>(X, meanb, invb, Cc, Nn, Bn, 1e-5f);
        int total = Bn * Cc * Nn;
        bn3d_apply_kernel<<<cdiv(total, 256), 256, 0, stream>>>(X, meanb, invb, g, bb, Cc, Nn, total, relu);
    };
    auto gemm = [&](const float* W, const float* X, const float* bias, float* Y, int O, int C) {
        dim3 g(Nn / 64, O / 64, Bn);
        wmma_gemm_kernel<<<g, 128, 0, stream>>>(W, X, bias, Y, O, C, Nn);
    };
    auto tnet = [&](int base, const float* Xin, int Cin, int kk, float* mOut) {
        if (Cin == 64) {
            gemm(P(base + 19), Xin, P(base + 0), bufA, 64, 64);
        } else {
            int tot = Bn * 64 * Nn;
            conv_small_kernel<<<cdiv(tot, 256), 256, 0, stream>>>(P(base + 19), Xin, P(base + 0), bufA, 64, Cin, Nn, Bn);
        }
        bn3d(bufA, 64, P(base + 4), P(base + 3), 1);
        gemm(P(base + 20), bufA, P(base + 1), bufB, 128, 64);
        bn3d(bufB, 128, P(base + 6), P(base + 5), 1);
        gemm(P(base + 21), bufB, P(base + 2), bufC, 1024, 128);
        bn3d(bufC, 1024, P(base + 8), P(base + 7), 1);
        maxn_kernel<<<Bn * 1024, 256, 0, stream>>>(bufC, pooled, Nn);
        fc_kernel<<<cdiv(Bn * 512, 256), 256, 0, stream>>>(P(base + 16), pooled, P(base + 13), f1, 512, 1024, Bn);
        bn1d_kernel<<<cdiv(512, 256), 256, 0, stream>>>(f1, P(base + 10), P(base + 9), 512, Bn, 1e-5f, 1);
        fc_kernel<<<cdiv(Bn * 256, 256), 256, 0, stream>>>(P(base + 17), f1, P(base + 14), f2, 256, 512, Bn);
        bn1d_kernel<<<1, 256, 0, stream>>>(f2, P(base + 12), P(base + 11), 256, Bn, 1e-5f, 1);
        int kk2 = kk * kk;
        fc_kernel<<<cdiv(Bn * kk2, 256), 256, 0, stream>>>(P(base + 18), f2, P(base + 15), f3, kk2, 256, Bn);
        addeye_kernel<<<cdiv(Bn * kk2, 256), 256, 0, stream>>>(f3, mOut, kk, Bn);
    };

    // ---- t3 TNet -> m3 ----
    tnet(13, x, 3, 3, m3);
    // xb3 = einsum('bcn,bcd->bdn', x, m3)
    transform_kernel<<<dim3(Nn / 256, 3, Bn), 256, 0, stream>>>(x, m3, xb3, 3, Nn);
    // normals from raw points
    normals_kernel<<<cdiv(Bn * Nn, 128), 128, 0, stream>>>(x, nrm, Nn, Bn);
    // concat -> (B,6,N)
    {
        int tot = Bn * 6 * Nn;
        concat6_kernel<<<cdiv(tot, 256), 256, 0, stream>>>(xb3, nrm, xb6, Nn, Bn);
    }
    // c1 (64 <- 6) + bn1 + relu
    {
        int tot = Bn * 64 * Nn;
        conv_small_kernel<<<cdiv(tot, 256), 256, 0, stream>>>(P(8), xb6, P(7), bufD, 64, 6, Nn, Bn);
    }
    bn3d(bufD, 64, P(2), P(1), 1);
    // ---- t64 TNet -> m64 ----
    tnet(35, bufD, 64, 64, m64);
    // xb = einsum('bcn,bcd->bdn', xb, m64)
    transform_kernel<<<dim3(Nn / 256, 64, Bn), 256, 0, stream>>>(bufD, m64, bufE, 64, Nn);
    // c2 (128 <- 64) + bn2 + relu
    gemm(P(10), bufE, P(9), bufB, 128, 64);
    bn3d(bufB, 128, P(4), P(3), 1);
    // c3 (1024 <- 128) + bn3 (no relu)
    gemm(P(12), bufB, P(11), bufC, 1024, 128);
    bn3d(bufC, 1024, P(6), P(5), 0);
    // global max pool -> out
    maxn_kernel<<<Bn * 1024, 256, 0, stream>>>(bufC, out_max, Nn);
}
